// GCN_Model_67216238182971
// MI455X (gfx1250) — matrix operand order repaired and checked by reading, not compiled, not run on hardware
//
#include <hip/hip_runtime.h>
#include <hip/hip_bf16.h>

typedef __attribute__((ext_vector_type(16))) _Float16 v16h;
typedef __attribute__((ext_vector_type(8)))  float    v8f;

// ---------------------------------------------------------------------------
// Degree / normalization
// ---------------------------------------------------------------------------
__global__ void deg_kernel(const int* __restrict__ dst, float* __restrict__ deg, int E) {
    int i = blockIdx.x * blockDim.x + threadIdx.x;
    if (i < E) atomicAdd(&deg[dst[i]], 1.0f);
}

__global__ void dinv_kernel(float* __restrict__ deg, int N) {
    int i = blockIdx.x * blockDim.x + threadIdx.x;
    if (i < N) deg[i] = rsqrtf(deg[i] + 1.0f);   // +1 self loop
}

// ---------------------------------------------------------------------------
// WMMA GEMM: H[M x NOUT] = X[M x K] * W[K x NOUT]   (f32 in, f16 MAC, f32 acc)
// One wave computes one 16x16 output tile. wave32; EXEC all-ones (uniform guard).
// ---------------------------------------------------------------------------
template <int K, int NOUT>
__global__ void gemm_wmma(const float* __restrict__ X, const float* __restrict__ W,
                          float* __restrict__ H, int M) {
    const int wave  = threadIdx.x >> 5;
    const int lane  = threadIdx.x & 31;
    const int mt    = blockIdx.x * (blockDim.x >> 5) + wave;  // 16-row tile index
    const int nt    = blockIdx.y;                             // 16-col tile index
    const int mtiles = M >> 4;                                // M % 16 == 0 here
    if (mt >= mtiles) return;                                 // wave-uniform

    const int r16  = lane & 15;   // row (A) / col (B,C,D) within tile
    const int half = lane >> 4;

    v8f acc = {};
    const float* xrow = X + (size_t)(mt * 16 + r16) * K;
    const int    col  = nt * 16 + r16;

#pragma unroll
    for (int k0 = 0; k0 < K; k0 += 32) {
        v16h a, b;
#pragma unroll
        for (int v = 0; v < 8; ++v) {
            // A fragment: 16-bit A 16x32 layout (VGPR v -> K pattern, lanes split M/halfK)
            int ka = k0 + ((v & 4) << 2) + half * 8 + (v & 3) * 2;
            float2 xa = *(const float2*)(xrow + ka);
            a[2 * v]     = (_Float16)xa.x;
            a[2 * v + 1] = (_Float16)xa.y;
            // B fragment: same K pattern as A (VGPR v -> K pattern, lanes split N/halfK)
            int kb = k0 + ((v & 4) << 2) + half * 8 + (v & 3) * 2;
            b[2 * v]     = (_Float16)W[(size_t)kb * NOUT + col];
            b[2 * v + 1] = (_Float16)W[(size_t)(kb + 1) * NOUT + col];
        }
        acc = __builtin_amdgcn_wmma_f32_16x16x32_f16(
            /*neg_a=*/false, a, /*neg_b=*/false, b,
            /*c_mod=*/(short)0, acc, /*reuse_a=*/false, /*reuse_b=*/false);
    }

    // C/D layout: VGPR v -> row M = v + half*8; lane%16 -> col
#pragma unroll
    for (int v = 0; v < 8; ++v) {
        int r = mt * 16 + v + half * 8;
        H[(size_t)r * NOUT + col] = acc[v];
    }
}

// ---------------------------------------------------------------------------
// Edge aggregation: AGG[dst] += H[src] * dinv[src]*dinv[dst]
// One thread per (edge, 4-float chunk); for D=128 a wave == one edge (coalesced).
// ---------------------------------------------------------------------------
template <int D>
__global__ void agg_kernel(const float* __restrict__ H, const int* __restrict__ src,
                           const int* __restrict__ dst, const float* __restrict__ dinv,
                           float* __restrict__ AGG, int E) {
    const int CH = D / 4;
    long long idx = (long long)blockIdx.x * blockDim.x + threadIdx.x;
    int e = (int)(idx / CH);
    int c = (int)(idx % CH);
    if (e >= E) return;
    int s = src[e], d = dst[e];
    float norm = dinv[s] * dinv[d];
    float4 hv = *(const float4*)(H + (size_t)s * D + c * 4);
    float* o = AGG + (size_t)d * D + c * 4;
    atomicAdd(o + 0, hv.x * norm);
    atomicAdd(o + 1, hv.y * norm);
    atomicAdd(o + 2, hv.z * norm);
    atomicAdd(o + 3, hv.w * norm);
}

// ---------------------------------------------------------------------------
// Epilogue: Z = act(AGG + H * dinv^2 + bias), written in place into AGG.
// ---------------------------------------------------------------------------
template <int D, bool RELU>
__global__ void finish_kernel(const float* __restrict__ H, const float* __restrict__ dinv,
                              const float* __restrict__ bias, float* __restrict__ AGG, int N) {
    const int CH = D / 4;
    long long idx = (long long)blockIdx.x * blockDim.x + threadIdx.x;
    int n = (int)(idx / CH);
    int c = (int)(idx % CH);
    if (n >= N) return;
    float di = dinv[n];
    float sl = di * di;
    const float4 h = *(const float4*)(H + (size_t)n * D + c * 4);
    float4 a = *(const float4*)(AGG + (size_t)n * D + c * 4);
    const float4 b = *(const float4*)(bias + c * 4);
    a.x += h.x * sl + b.x;
    a.y += h.y * sl + b.y;
    a.z += h.z * sl + b.z;
    a.w += h.w * sl + b.w;
    if (RELU) {
        a.x = fmaxf(a.x, 0.0f); a.y = fmaxf(a.y, 0.0f);
        a.z = fmaxf(a.z, 0.0f); a.w = fmaxf(a.w, 0.0f);
    }
    *(float4*)(AGG + (size_t)n * D + c * 4) = a;
}

// ---------------------------------------------------------------------------
// Fused head: out = relu(Z3 @ W4 + b4) @ W5 + b5    (16 -> 8 -> 2)
// ---------------------------------------------------------------------------
__global__ void head_kernel(const float* __restrict__ Z3,
                            const float* __restrict__ W4, const float* __restrict__ b4,
                            const float* __restrict__ W5, const float* __restrict__ b5,
                            float* __restrict__ out, int N) {
    __shared__ float sW4[16 * 8];
    __shared__ float sb4[8];
    __shared__ float sW5[8 * 2];
    __shared__ float sb5[2];
    int t = threadIdx.x;
    if (t < 128) sW4[t] = W4[t];
    if (t < 8)   sb4[t] = b4[t];
    if (t < 16)  sW5[t] = W5[t];
    if (t < 2)   sb5[t] = b5[t];
    __syncthreads();

    int n = blockIdx.x * blockDim.x + t;
    if (n >= N) return;

    float z[16];
    const float4* zp = (const float4*)(Z3 + (size_t)n * 16);
#pragma unroll
    for (int q = 0; q < 4; ++q) {
        float4 v = zp[q];
        z[4 * q + 0] = v.x; z[4 * q + 1] = v.y; z[4 * q + 2] = v.z; z[4 * q + 3] = v.w;
    }
    float h4[8];
#pragma unroll
    for (int j = 0; j < 8; ++j) {
        float acc = sb4[j];
#pragma unroll
        for (int k = 0; k < 16; ++k) acc += z[k] * sW4[k * 8 + j];
        h4[j] = fmaxf(acc, 0.0f);
    }
#pragma unroll
    for (int o = 0; o < 2; ++o) {
        float acc = sb5[o];
#pragma unroll
        for (int j = 0; j < 8; ++j) acc += h4[j] * sW5[j * 2 + o];
        out[(size_t)n * 2 + o] = acc;
    }
}

// ---------------------------------------------------------------------------
// Launch
// ---------------------------------------------------------------------------
extern "C" void kernel_launch(void* const* d_in, const int* in_sizes, int n_in,
                              void* d_out, int out_size, void* d_ws, size_t ws_size,
                              hipStream_t stream) {
    const float* x  = (const float*)d_in[0];
    const int*   ei = (const int*)d_in[1];
    const float* W1 = (const float*)d_in[2];  const float* b1 = (const float*)d_in[3];
    const float* W2 = (const float*)d_in[4];  const float* b2 = (const float*)d_in[5];
    const float* W3 = (const float*)d_in[6];  const float* b3 = (const float*)d_in[7];
    const float* W4 = (const float*)d_in[8];  const float* b4 = (const float*)d_in[9];
    const float* W5 = (const float*)d_in[10]; const float* b5 = (const float*)d_in[11];
    float* out = (float*)d_out;

    const int N = in_sizes[0] / 128;   // 100000
    const int E = in_sizes[1] / 2;     // 3200000
    const int* src = ei;
    const int* dst = ei + E;

    // Workspace layout: dinv[N] | R0[N*128] | R1[N*128] | R2[N*128]
    float* dinv = (float*)d_ws;
    size_t dpad = ((size_t)N + 63) & ~(size_t)63;
    float* R0 = dinv + dpad;
    float* R1 = R0 + (size_t)N * 128;
    float* R2 = R1 + (size_t)N * 128;

    const int BT = 256;
    const int WAVES = BT / 32;

    // --- degrees -> dinv
    hipMemsetAsync(dinv, 0, (size_t)N * sizeof(float), stream);
    deg_kernel<<<(E + BT - 1) / BT, BT, 0, stream>>>(dst, dinv, E);
    dinv_kernel<<<(N + BT - 1) / BT, BT, 0, stream>>>(dinv, N);

    const int mtiles = N / 16;
    const int gx = (mtiles + WAVES - 1) / WAVES;

    // --- Layer 1: 128 -> 128, relu
    gemm_wmma<128, 128><<<dim3(gx, 128 / 16), BT, 0, stream>>>(x, W1, R0, N);
    hipMemsetAsync(R1, 0, (size_t)N * 128 * sizeof(float), stream);
    {
        long long tot = (long long)E * (128 / 4);
        agg_kernel<128><<<(unsigned)((tot + BT - 1) / BT), BT, 0, stream>>>(R0, src, dst, dinv, R1, E);
        long long ft = (long long)N * (128 / 4);
        finish_kernel<128, true><<<(unsigned)((ft + BT - 1) / BT), BT, 0, stream>>>(R0, dinv, b1, R1, N);
    }

    // --- Layer 2: 128 -> 32, relu
    gemm_wmma<128, 32><<<dim3(gx, 32 / 16), BT, 0, stream>>>(R1, W2, R2, N);
    hipMemsetAsync(R0, 0, (size_t)N * 32 * sizeof(float), stream);
    {
        long long tot = (long long)E * (32 / 4);
        agg_kernel<32><<<(unsigned)((tot + BT - 1) / BT), BT, 0, stream>>>(R2, src, dst, dinv, R0, E);
        long long ft = (long long)N * (32 / 4);
        finish_kernel<32, true><<<(unsigned)((ft + BT - 1) / BT), BT, 0, stream>>>(R2, dinv, b2, R0, N);
    }

    // --- Layer 3: 32 -> 16, relu
    gemm_wmma<32, 16><<<dim3(gx, 16 / 16), BT, 0, stream>>>(R0, W3, R1, N);
    hipMemsetAsync(R2, 0, (size_t)N * 16 * sizeof(float), stream);
    {
        long long tot = (long long)E * (16 / 4);
        agg_kernel<16><<<(unsigned)((tot + BT - 1) / BT), BT, 0, stream>>>(R1, src, dst, dinv, R2, E);
        long long ft = (long long)N * (16 / 4);
        finish_kernel<16, true><<<(unsigned)((ft + BT - 1) / BT), BT, 0, stream>>>(R1, dinv, b3, R2, N);
    }

    // --- Head: 16 -> 8 -> 2
    head_kernel<<<(N + BT - 1) / BT, BT, 0, stream>>>(R2, W4, b4, W5, b5, out, N);
}
